// QuantLayer_10866267259536
// MI455X (gfx1250) — matrix operand order repaired and checked by reading, not compiled, not run on hardware
//
#include <hip/hip_runtime.h>
#include <hip/hip_bf16.h>

// ---------------------------------------------------------------------------
// QuantLayer on MI455X (gfx1250).
//   out[t] = b_post + sum_g M[g, argmax_v(logits[t,g,v]+gumbel), :]
// with M[g,v,:] = codebook[g,v,:] @ W_post[g*64:(g+1)*64,:]   (precomputed)
// and  logits = (x @ W_pre + b_pre) @ W_proj + b_proj,
// x@W_pre done with v_wmma_f32_16x16x32_bf16; M table staged into LDS by the
// Tensor Data Mover (tensor_load_to_lds), overlapped with the WMMA phase.
// ---------------------------------------------------------------------------

typedef __attribute__((ext_vector_type(16))) __bf16          v16bf;
typedef __attribute__((ext_vector_type(8)))  float           v8f;
typedef __attribute__((ext_vector_type(16))) unsigned short  v16us;
typedef __attribute__((ext_vector_type(8)))  unsigned short  v8us;
typedef __attribute__((ext_vector_type(4)))  unsigned int    v4ui;
typedef __attribute__((ext_vector_type(8)))  int             v8si;
typedef __attribute__((ext_vector_type(4)))  int             v4si;

#define TB        64        // tokens per block
#define NTHREADS  256       // 8 wave32
#define IN_DIM    512
#define QDIM      32
#define GV        64        // G*V
#define OUT_DIM   512
#define HS_LD     36        // padded leading dim for h in LDS

// LDS layout (bytes)
#define SM_WT_OFF    0                         // W_pre^T bf16 [32][512]  = 32768
#define SM_XS_OFF    (SM_WT_OFF + 32768)       // x tile bf16  [64][512]  = 65536
#define SM_ML_OFF    (SM_XS_OFF + 65536)       // M table f32  [64][512]  = 131072
#define SM_WPJ_OFF   (SM_ML_OFF + 131072)      // W_proj f32   [32][64]   = 8192
#define SM_BPOST_OFF (SM_WPJ_OFF + 8192)       // b_post f32   [512]      = 2048
#define SM_BPRE_OFF  (SM_BPOST_OFF + 2048)     // b_pre f32    [32]       = 128
#define SM_HS_OFF    (SM_BPRE_OFF + 128)       // h f32        [64][36]   = 9216
#define SM_IDX_OFF   (SM_HS_OFF + 9216)        // idx u8       [64][8]    = 512
#define SM_TOTAL     (SM_IDX_OFF + 512)        // = 249472 bytes

#if __has_builtin(__builtin_amdgcn_tensor_load_to_lds) && \
    __has_builtin(__builtin_amdgcn_s_wait_tensorcnt)
#define USE_TDM 1
#else
#define USE_TDM 0
#endif

static __device__ __forceinline__ unsigned short f2bf(float f) {
    unsigned int b = __builtin_bit_cast(unsigned int, f);
    b += 0x7fffu + ((b >> 16) & 1u);           // round-to-nearest-even
    return (unsigned short)(b >> 16);
}

// ---------------------------------------------------------------------------
// Kernel 1: M[g*8+v][c] = sum_d codebook[g][v][d] * W_post[(g*64+d)*512 + c]
// grid = 64 blocks (one per (g,v)), 256 threads (2 columns each).
// ---------------------------------------------------------------------------
__global__ __launch_bounds__(NTHREADS) void qlayer_mtab(
    const float* __restrict__ codebook,        // (8,8,64)
    const float* __restrict__ W_post,          // (512,512)
    float* __restrict__ Mtab)                  // (64,512) scratch
{
    __shared__ float cb[64];
    const int gv = blockIdx.x;                 // 0..63
    const int tid = threadIdx.x;
    if (tid < 64) cb[tid] = codebook[gv * 64 + tid];
    __syncthreads();

    const int g = gv >> 3;
    float a0 = 0.f, a1 = 0.f;
#pragma unroll 8
    for (int d = 0; d < 64; ++d) {
        const float w = cb[d];
        const float* wp = W_post + (size_t)(g * 64 + d) * OUT_DIM;
        a0 = fmaf(w, wp[tid],       a0);
        a1 = fmaf(w, wp[tid + 256], a1);
    }
    Mtab[gv * OUT_DIM + tid]       = a0;
    Mtab[gv * OUT_DIM + tid + 256] = a1;
}

// ---------------------------------------------------------------------------
// Kernel 2: main fused kernel. One block = 64 tokens, 256 threads.
// ---------------------------------------------------------------------------
__global__ __launch_bounds__(NTHREADS) void qlayer_main(
    const float* __restrict__ x,               // (B,T,512)
    const float* __restrict__ W_pre,           // (512,32)
    const float* __restrict__ b_pre,           // (32)
    const float* __restrict__ W_proj,          // (32,64)
    const float* __restrict__ b_proj,          // (64)
    const float* __restrict__ Mtab,            // (64,512) precomputed
    const float* __restrict__ b_post,          // (512)
    const float* __restrict__ u,               // (B,T,8,8)
    float* __restrict__ out)                   // (B,T,512)
{
    extern __shared__ char smem[];
    unsigned short* Wt    = (unsigned short*)(smem + SM_WT_OFF);
    unsigned short* xs    = (unsigned short*)(smem + SM_XS_OFF);
    float*          Ml    = (float*)(smem + SM_ML_OFF);
    float*          Wpj   = (float*)(smem + SM_WPJ_OFF);
    float*          bpost = (float*)(smem + SM_BPOST_OFF);
    float*          bpre  = (float*)(smem + SM_BPRE_OFF);
    float*          hs    = (float*)(smem + SM_HS_OFF);
    unsigned char*  idxs  = (unsigned char*)(smem + SM_IDX_OFF);

    const int tid  = threadIdx.x;
    const int tok0 = blockIdx.x * TB;

    // Prefetch the gumbel-noise tile for this block (global_prefetch_b8).
    if (tid < 128)
        __builtin_prefetch((const char*)(u + (size_t)tok0 * GV) + tid * 128, 0, 1);

#if USE_TDM
    // ---- Kick off the M-table copy (128 KB, global L2 -> LDS) on the TDM.
    // D# per CDNA5 ISA: 2D tensor, f32 elements,
    // tile = 512 x 64 rows, stride 512. Runs concurrently with phases A-C;
    // waited on (TENSORcnt) just before phase D.
    if (tid < 32) {                            // wave 0 only
        const unsigned long long ga = (unsigned long long)(uintptr_t)Mtab;
        v4ui g0;
        g0[0] = 1u;                                    // count=1 (valid user D#)
        g0[1] = (unsigned)SM_ML_OFF;                   // lds_addr (bytes)
        g0[2] = (unsigned)(ga & 0xffffffffull);        // global_addr[31:0]
        g0[3] = (unsigned)((ga >> 32) & 0x01ffffffull) // global_addr[56:32]
                | 0x80000000u;                         // type=2 ("image")
        v8si g1;
        g1[0] = (int)(2u << 16);                       // wg_mask=0, data_size=4B
        g1[1] = (int)(512u << 16);                     // tensor_dim0 = 512
        g1[2] = (int)(64u  << 16);                     // tensor_dim1 = 64
        g1[3] = (int)(512u << 16);                     // tile_dim0 = 512
        g1[4] = 64;                                    // tile_dim1 = 64
        g1[5] = 512;                                   // tensor_dim0_stride = 512
        g1[6] = 0;
        g1[7] = 0;
        v4si gz = {0, 0, 0, 0};
#if __has_include(<hip/amd_detail/amd_gfx1250_TDM.h>)
        v8si gz8 = {0, 0, 0, 0, 0, 0, 0, 0};           // clang-23 6-arg form
        __builtin_amdgcn_tensor_load_to_lds(g0, g1, gz, gz, gz8, 0);
#else
        __builtin_amdgcn_tensor_load_to_lds(g0, g1, gz, gz, 0);  // ROCm 7.2
#endif
    }
#endif

    // ---- Phase A: stage everything else into LDS -------------------------
    // W_pre (512x32, row-major k,n) -> Wt[n][k] bf16 (transposed).
    for (int i = tid; i < IN_DIM * QDIM; i += NTHREADS) {
        const int k = i >> 5, n = i & 31;
        Wt[n * IN_DIM + k] = f2bf(W_pre[i]);
    }
    // x tile (64x512 f32) -> xs bf16, float4-vectorized global reads.
    {
        const float4* xg = (const float4*)(x + (size_t)tok0 * IN_DIM);
        for (int i = tid; i < TB * IN_DIM / 4; i += NTHREADS) {
            const float4 v = xg[i];
            const int b = i * 4;
            xs[b + 0] = f2bf(v.x); xs[b + 1] = f2bf(v.y);
            xs[b + 2] = f2bf(v.z); xs[b + 3] = f2bf(v.w);
        }
    }
#if !USE_TDM
    // Fallback: M table (64x512 f32) via vector loads.
    {
        const float4* mg = (const float4*)Mtab;
        float4* ml4 = (float4*)Ml;
        for (int i = tid; i < GV * OUT_DIM / 4; i += NTHREADS) ml4[i] = mg[i];
    }
#endif
    for (int i = tid; i < QDIM * GV; i += NTHREADS) Wpj[i] = W_proj[i];
    for (int i = tid; i < OUT_DIM; i += NTHREADS)   bpost[i] = b_post[i];
    if (tid < QDIM) bpre[tid] = b_pre[tid];
    __syncthreads();

    // ---- Phase B: h = x @ W_pre via bf16 WMMA ----------------------------
    // 8 waves; wave w owns 16x16 tile: rows rt*16..+15, cols n0..n0+15.
    {
        const int wave = tid >> 5, lane = tid & 31;
        const int rt   = wave >> 1;
        const int n0   = (wave & 1) * 16;
        const int mrow = rt * 16 + (lane & 15);
        const int alo  = (lane < 16) ? 0 : 16;   // A: low lanes K0-15, high K16-31
        const int blo  = (lane < 16) ? 0 : 16;   // B: low lanes K0-15, high K16-31
        const int bn   = n0 + (lane & 15);

        v8f acc = {};
#pragma unroll
        for (int kk = 0; kk < 16; ++kk) {
            const int k0 = kk * 32;
            const v8us a_lo = *(const v8us*)(xs + mrow * IN_DIM + k0 + alo);
            const v8us a_hi = *(const v8us*)(xs + mrow * IN_DIM + k0 + alo + 8);
            v16us au;
#pragma unroll
            for (int i = 0; i < 8; ++i) { au[i] = a_lo[i]; au[i + 8] = a_hi[i]; }
            const v16us bu = *(const v16us*)(Wt + bn * IN_DIM + k0 + blo);

            const v16bf av = __builtin_bit_cast(v16bf, au);
            const v16bf bv = __builtin_bit_cast(v16bf, bu);
            acc = __builtin_amdgcn_wmma_f32_16x16x32_bf16(
                false, av, false, bv, (short)0, acc, false, false);
        }
        // C layout: VGPR r -> row (r | r+8), col = lane&15 (+n0).
        const int cr = rt * 16 + ((lane < 16) ? 0 : 8);
        const int cc = n0 + (lane & 15);
#pragma unroll
        for (int r = 0; r < 8; ++r)
            hs[(cr + r) * HS_LD + cc] = acc[r] + bpre[cc];
    }
    __syncthreads();

    // ---- Phase C: logits + gumbel + per-group argmax ---------------------
#pragma unroll
    for (int p = 0; p < 2; ++p) {
        const int pi = tid + p * NTHREADS;     // 0..511 = (token, group)
        const int t  = pi >> 3;
        const int g  = pi & 7;

        float hr[QDIM];
#pragma unroll
        for (int j = 0; j < QDIM; ++j) hr[j] = hs[t * HS_LD + j];

        const float* ug = u + ((size_t)(tok0 + t)) * GV + g * 8;
        float best = -1e30f; int bi = 0;
#pragma unroll
        for (int v = 0; v < 8; ++v) {
            float lg = b_proj[g * 8 + v];
#pragma unroll
            for (int j = 0; j < QDIM; ++j)
                lg = fmaf(hr[j], Wpj[j * GV + g * 8 + v], lg);
            const float gum = -__logf(-__logf(ug[v] + 1e-10f) + 1e-10f);
            const float val = lg + gum;        // argmax invariant under /tau+softmax
            if (val > best) { best = val; bi = v; }
        }
        idxs[t * 8 + g] = (unsigned char)bi;
    }

#if USE_TDM
    if (tid < 32) __builtin_amdgcn_s_wait_tensorcnt(0);   // M table landed in LDS
#endif
    __syncthreads();

    // ---- Phase D: out[t] = b_post + sum_g M[g*8+idx] ---------------------
    {
        const int c = tid * 2;                 // each thread: 2 adjacent columns
        const float bp0 = bpost[c], bp1 = bpost[c + 1];
        for (int t = 0; t < TB; ++t) {
            const unsigned int iw0 = *(const unsigned int*)(idxs + t * 8);
            const unsigned int iw1 = *(const unsigned int*)(idxs + t * 8 + 4);
            float o0 = bp0, o1 = bp1;
#pragma unroll
            for (int g = 0; g < 8; ++g) {
                const unsigned int v =
                    ((g < 4) ? (iw0 >> (8 * g)) : (iw1 >> (8 * (g - 4)))) & 7u;
                const float* mr = Ml + (g * 8 + v) * OUT_DIM + c;
                o0 += mr[0]; o1 += mr[1];
            }
            float2 ov; ov.x = o0; ov.y = o1;
            *(float2*)(out + ((size_t)(tok0 + t)) * OUT_DIM + c) = ov;
        }
    }
}

// ---------------------------------------------------------------------------
extern "C" void kernel_launch(void* const* d_in, const int* in_sizes, int n_in,
                              void* d_out, int out_size, void* d_ws, size_t ws_size,
                              hipStream_t stream) {
    const float* x        = (const float*)d_in[0];
    const float* W_pre    = (const float*)d_in[1];
    const float* b_pre    = (const float*)d_in[2];
    const float* W_proj   = (const float*)d_in[3];
    const float* b_proj   = (const float*)d_in[4];
    const float* codebook = (const float*)d_in[5];
    const float* W_post   = (const float*)d_in[6];
    const float* b_post   = (const float*)d_in[7];
    const float* u        = (const float*)d_in[8];
    float* out  = (float*)d_out;
    float* Mtab = (float*)d_ws;                // 64*512*4 = 128 KB scratch

    (void)in_sizes; (void)n_in; (void)out_size; (void)ws_size;

    hipFuncSetAttribute((const void*)qlayer_main,
                        hipFuncAttributeMaxDynamicSharedMemorySize,
                        (int)SM_TOTAL);

    qlayer_mtab<<<GV, NTHREADS, 0, stream>>>(codebook, W_post, Mtab);

    const int n_tokens = 4 * 8192;
    qlayer_main<<<n_tokens / TB, NTHREADS, SM_TOTAL, stream>>>(
        x, W_pre, b_pre, W_proj, b_proj, Mtab, b_post, u, out);
}
